// MultiHeadedAttention_77094662963343
// MI455X (gfx1250) — matrix operand order repaired and checked by reading, not compiled, not run on hardware
//
#include <hip/hip_runtime.h>

#define D_MODEL 1024
#define SEQ_N   2048
#define NHEAD   16
#define HDIM    64
#define NBATCH  2

#if __has_builtin(__builtin_amdgcn_tensor_load_to_lds)
#define USE_TDM 1
#else
#define USE_TDM 0
#endif

typedef __attribute__((ext_vector_type(16))) __bf16 v16bf;
typedef __attribute__((ext_vector_type(8)))  float  v8f;
typedef __attribute__((ext_vector_type(4)))  unsigned int v4u;
typedef __attribute__((ext_vector_type(8)))  int v8i;
typedef __attribute__((ext_vector_type(4)))  int v4i;

union AF { v16bf v; unsigned int u[8]; v4u q[2]; };
union CF { v8f   v; float        f[8]; };

__device__ __forceinline__ unsigned short f2bf(float x) {
  unsigned int u = __float_as_uint(x);
  u += 0x7FFFu + ((u >> 16) & 1u);          // round-to-nearest-even truncate
  return (unsigned short)(u >> 16);
}

#if USE_TDM
// TDM 2D tile load: tile_x contiguous elements (2B) x tile_y rows at
// stride_e elements, from gptr into contiguous LDS at lds_byte ([y][x]).
__device__ __forceinline__ void tdm_load_2d_bf16(
    const unsigned short* gptr, unsigned lds_byte,
    unsigned tile_x, unsigned tile_y,
    unsigned long long stride_e, unsigned tensor_x, unsigned tensor_y)
{
  unsigned long long ga = (unsigned long long)(uintptr_t)gptr;
  v4u g0;
  g0.x = 1u;                                        // count=1, user-mode
  g0.y = lds_byte;                                  // lds_addr @ [63:32]
  g0.z = (unsigned)(ga & 0xFFFFFFFFu);              // global_addr[31:0]
  g0.w = (unsigned)((ga >> 32) & 0x01FFFFFFu) | (2u << 30);  // ga[56:32] | type=2
  v8i g1;
  g1[0] = (int)(1u << 16);                          // data_size=1 (2 bytes)
  g1[1] = (int)((tensor_x & 0xFFFFu) << 16);        // tensor_dim0[15:0]
  g1[2] = (int)(((tensor_x >> 16) & 0xFFFFu) |
                ((tensor_y & 0xFFFFu) << 16));      // tensor_dim1[15:0]
  g1[3] = (int)(((tensor_y >> 16) & 0xFFFFu) |
                ((tile_x & 0xFFFFu) << 16));        // tile_dim0
  g1[4] = (int)(tile_y & 0xFFFFu);                  // tile_dim1 (tile_dim2=0)
  g1[5] = (int)(unsigned)(stride_e & 0xFFFFFFFFu);  // dim0_stride[31:0]
  g1[6] = (int)(unsigned)((stride_e >> 32) & 0xFFFFu);
  g1[7] = 0;
  v4i gz = {0, 0, 0, 0};
#if __clang_major__ >= 23
  v8i gz8 = {0, 0, 0, 0, 0, 0, 0, 0};
  __builtin_amdgcn_tensor_load_to_lds(g0, g1, gz, gz, gz8, 0);
#else
  __builtin_amdgcn_tensor_load_to_lds(g0, g1, gz, gz, 0);
#endif
}
#endif

// ---------------------------------------------------------------------------
// C[m0:m0+64, n0:n0+64] = W x X + bias  (per batch z)
// ---------------------------------------------------------------------------
template <bool BF16OUT>
__global__ __launch_bounds__(128) void proj_wmma(
    const float* __restrict__ W, const float* __restrict__ Xin,
    const float* __restrict__ bias, void* __restrict__ outv)
{
  __shared__ __align__(16) unsigned short sW[64][32];   // [m][k]
  __shared__ __align__(16) unsigned short sX[64][32];   // [n][k]

  const int tid  = threadIdx.x;
  const int lane = tid & 31;
  const int wave = tid >> 5;
  const int n0 = blockIdx.x * 64;
  const int m0 = blockIdx.y * 64;
  const int b  = blockIdx.z;
  const float* X = Xin + (size_t)b * D_MODEL * SEQ_N;

  CF acc[4];
#pragma unroll
  for (int g = 0; g < 4; ++g)
#pragma unroll
    for (int e = 0; e < 8; ++e) acc[g].f[e] = 0.0f;

  for (int k0 = 0; k0 < D_MODEL; k0 += 32) {
    {
      const int c = tid & 31, r0 = tid >> 5;      // 4 rows / pass
#pragma unroll
      for (int p = 0; p < 16; ++p) {
        const int r = r0 + 4 * p;
        sW[r][c] = f2bf(W[(size_t)(m0 + r) * D_MODEL + (k0 + c)]);
      }
    }
    {
      const int c = tid & 63, r0 = tid >> 6;      // 2 k-rows / pass
#pragma unroll
      for (int p = 0; p < 16; ++p) {
        const int r = r0 + 2 * p;
        sX[c][r] = f2bf(X[(size_t)(k0 + r) * SEQ_N + (n0 + c)]);
      }
    }
    if (k0 + 32 < D_MODEL) {   // prefetch next K-panel behind the WMMAs
      __builtin_prefetch(&W[(size_t)(m0 + (tid >> 5)) * D_MODEL + (k0 + 32 + (tid & 31))], 0, 3);
      __builtin_prefetch(&X[(size_t)(k0 + 32 + (tid >> 6)) * SEQ_N + (n0 + (tid & 63))], 0, 3);
    }
    __syncthreads();

    AF a;
    const int mrow = (wave << 4) + (lane & 15);
    {
      const unsigned short* ab = &sW[mrow][(lane >> 4) << 3];
      a.q[0] = *(const v4u*)ab;            // K 0..7   (+8 for hi half-lanes)
      a.q[1] = *(const v4u*)(ab + 16);     // K 16..23
    }
#pragma unroll
    for (int g = 0; g < 4; ++g) {
      AF bfr;
      const int nc = (g << 4) + (lane & 15);
      const unsigned short* bb = &sX[nc][(lane >> 4) << 3];
      bfr.q[0] = *(const v4u*)bb;          // K 0..7   (+8 for hi half-lanes)
      bfr.q[1] = *(const v4u*)(bb + 16);   // K 16..23
      acc[g].v = __builtin_amdgcn_wmma_f32_16x16x32_bf16(
          false, a.v, false, bfr.v, (short)0, acc[g].v, false, false);
    }
    __syncthreads();
  }

  const int rbase = m0 + (wave << 4) + ((lane >> 4) << 3);
  const size_t ob = (size_t)b * D_MODEL * SEQ_N;
#pragma unroll
  for (int e = 0; e < 8; ++e) {
    const int r = rbase + e;
    const float bv = bias[r];
#pragma unroll
    for (int g = 0; g < 4; ++g) {
      const int c = n0 + (g << 4) + (lane & 15);
      const float val = acc[g].f[e] + bv;
      if (BF16OUT) ((unsigned short*)outv)[ob + (size_t)r * SEQ_N + c] = f2bf(val);
      else         ((float*)outv)[ob + (size_t)r * SEQ_N + c] = val;
    }
  }
}

// ---------------------------------------------------------------------------
// Flash attention per (b, h, 64-query-row tile). Channel c = d*NHEAD + h.
// K/V tiles staged by the TDM (double-buffered raw), then cooperatively
// repacked so every WMMA fragment read is an aligned ds_load_b128.
// Packed layout: dword [x][dp] = ( T[2dp][x], T[2dp+1][x] ).
// ---------------------------------------------------------------------------
__global__ __launch_bounds__(128) void attn_wmma(
    const unsigned short* __restrict__ Qp, const unsigned short* __restrict__ Kp,
    const unsigned short* __restrict__ Vp, float* __restrict__ Xout)
{
#if USE_TDM
  __shared__ __align__(16) unsigned short sQd[64][64];      // raw [d][n]
  __shared__ __align__(16) unsigned short sKd[2][64][64];   // raw ping-pong [d][m]
#endif
  __shared__ __align__(16) unsigned int   sQq[64][32];      // packed [n][dp]
  __shared__ __align__(16) unsigned int   sKq[64][32];      // packed [m][dp]
  __shared__ __align__(16) unsigned short sVd[2][64][64];   // raw [d][m]
  __shared__ __align__(16) unsigned short sP[4][16][64];    // per-wave P (C->A)

  const int tid  = threadIdx.x;
  const int lane = tid & 31;
  const int wave = tid >> 5;
  const int n0 = blockIdx.x * 64;
  const int h  = blockIdx.y;
  const int b  = blockIdx.z;
  const size_t base = (size_t)b * D_MODEL * SEQ_N;
  const size_t hb   = base + (size_t)h * SEQ_N;
  const unsigned long long rowstride = (unsigned long long)NHEAD * SEQ_N;
  const int NT = SEQ_N / 64;

#if USE_TDM
  const unsigned ldsQ = (unsigned)(unsigned long long)(uintptr_t)(const void*)&sQd[0][0];
  unsigned ldsK[2], ldsV[2];
  ldsK[0] = (unsigned)(unsigned long long)(uintptr_t)(const void*)&sKd[0][0][0];
  ldsK[1] = (unsigned)(unsigned long long)(uintptr_t)(const void*)&sKd[1][0][0];
  ldsV[0] = (unsigned)(unsigned long long)(uintptr_t)(const void*)&sVd[0][0][0];
  ldsV[1] = (unsigned)(unsigned long long)(uintptr_t)(const void*)&sVd[1][0][0];
  if (wave == 0) {
    tdm_load_2d_bf16(Qp + hb + n0, ldsQ, 64, 64, rowstride, SEQ_N, D_MODEL);
    tdm_load_2d_bf16(Kp + hb, ldsK[0], 64, 64, rowstride, SEQ_N, D_MODEL);
    tdm_load_2d_bf16(Vp + hb, ldsV[0], 64, 64, rowstride, SEQ_N, D_MODEL);
    __builtin_amdgcn_s_wait_tensorcnt(0);
  }
  __syncthreads();
  { // repack Q: sQd[d][n] -> sQq[n][dp]
    const int x2 = (tid & 31) * 2, dp0 = tid >> 5;
#pragma unroll
    for (int p = 0; p < 8; ++p) {
      const int dp = dp0 + 4 * p;
      const unsigned a  = *(const unsigned*)&sQd[2 * dp][x2];
      const unsigned b2 = *(const unsigned*)&sQd[2 * dp + 1][x2];
      sQq[x2][dp]     = (a & 0xFFFFu) | (b2 << 16);
      sQq[x2 + 1][dp] = (a >> 16) | (b2 & 0xFFFF0000u);
    }
  }
  __syncthreads();
#else
  { // stage Q directly packed from global
    const int n = tid & 63, x0 = tid >> 6;
    for (int p = 0; p < 16; ++p) {
      const int dp = x0 + 2 * p;
      const unsigned lo = Qp[hb + (size_t)(2 * dp) * rowstride + n0 + n];
      const unsigned hi = Qp[hb + (size_t)(2 * dp + 1) * rowstride + n0 + n];
      sQq[n][dp] = lo | (hi << 16);
    }
  }
  __syncthreads();
#endif

  AF qa[2];
  const int qrow = (wave << 4) + (lane & 15);
#pragma unroll
  for (int c = 0; c < 2; ++c) {
    const int dq = 16 * c + ((lane >> 4) << 2);
    qa[c].q[0] = *(const v4u*)&sQq[qrow][dq];       // dwords dq..dq+3
    qa[c].q[1] = *(const v4u*)&sQq[qrow][dq + 8];   // dwords dq+8..dq+11
  }

  CF acc[4];
  float row_m[8], row_l[8];
#pragma unroll
  for (int g = 0; g < 4; ++g)
#pragma unroll
    for (int e = 0; e < 8; ++e) acc[g].f[e] = 0.0f;
#pragma unroll
  for (int e = 0; e < 8; ++e) { row_m[e] = -3.0e38f; row_l[e] = 0.0f; }

  for (int mt = 0; mt < NT; ++mt) {
    const int cur = mt & 1;
#if USE_TDM
    if (wave == 0 && (mt + 1) < NT) {   // DMA next tile behind this one
      const int nxt = 1 - cur;
      tdm_load_2d_bf16(Kp + hb + (mt + 1) * 64, ldsK[nxt], 64, 64, rowstride, SEQ_N, D_MODEL);
      tdm_load_2d_bf16(Vp + hb + (mt + 1) * 64, ldsV[nxt], 64, 64, rowstride, SEQ_N, D_MODEL);
    }
    { // repack K: sKd[cur][d][m] -> sKq[m][dp]
      const unsigned short (*Kr)[64] = sKd[cur];
      const int x2 = (tid & 31) * 2, dp0 = tid >> 5;
#pragma unroll
      for (int p = 0; p < 8; ++p) {
        const int dp = dp0 + 4 * p;
        const unsigned a  = *(const unsigned*)&Kr[2 * dp][x2];
        const unsigned b2 = *(const unsigned*)&Kr[2 * dp + 1][x2];
        sKq[x2][dp]     = (a & 0xFFFFu) | (b2 << 16);
        sKq[x2 + 1][dp] = (a >> 16) | (b2 & 0xFFFF0000u);
      }
    }
    __syncthreads();   // packed K ready
#else
    {
      const int mm0 = mt * 64;
      const int m = tid & 63, x0 = tid >> 6;
      for (int p = 0; p < 16; ++p) {
        const int dp = x0 + 2 * p;
        const unsigned lo = Kp[hb + (size_t)(2 * dp) * rowstride + mm0 + m];
        const unsigned hi = Kp[hb + (size_t)(2 * dp + 1) * rowstride + mm0 + m];
        sKq[m][dp] = lo | (hi << 16);
      }
      for (int p = 0; p < 32; ++p) {
        const int d = x0 + 2 * p;
        sVd[cur][d][m] = Vp[hb + (size_t)d * rowstride + mm0 + m];
      }
    }
    __syncthreads();
#endif
    const unsigned short (*Vt)[64] = sVd[cur];

    CF s[4];
#pragma unroll
    for (int g = 0; g < 4; ++g)
#pragma unroll
      for (int e = 0; e < 8; ++e) s[g].f[e] = 0.0f;

#pragma unroll
    for (int c = 0; c < 2; ++c)
#pragma unroll
      for (int g = 0; g < 4; ++g) {
        AF bfr;
        const int nc = (g << 4) + (lane & 15);               // m column
        const int dp = 16 * c + ((lane >> 4) << 2);          // d-pair base
        bfr.q[0] = *(const v4u*)&sKq[nc][dp];
        bfr.q[1] = *(const v4u*)&sKq[nc][dp + 8];
        s[g].v = __builtin_amdgcn_wmma_f32_16x16x32_bf16(
            false, qa[c].v, false, bfr.v, (short)0, s[g].v, false, false);
      }

#pragma unroll
    for (int g = 0; g < 4; ++g)
#pragma unroll
      for (int e = 0; e < 8; ++e) s[g].f[e] *= 0.125f;   // 1/sqrt(64)

    // online softmax: rows live across the 16-lane half (xor masks 1,2,4,8)
#pragma unroll
    for (int e = 0; e < 8; ++e) {
      float t = fmaxf(fmaxf(s[0].f[e], s[1].f[e]), fmaxf(s[2].f[e], s[3].f[e]));
      t = fmaxf(t, __shfl_xor(t, 1, 32));
      t = fmaxf(t, __shfl_xor(t, 2, 32));
      t = fmaxf(t, __shfl_xor(t, 4, 32));
      t = fmaxf(t, __shfl_xor(t, 8, 32));
      const float nm = fmaxf(row_m[e], t);
      const float corr = __expf(row_m[e] - nm);
      row_m[e] = nm;
      float sum = 0.0f;
#pragma unroll
      for (int g = 0; g < 4; ++g) {
        const float p = __expf(s[g].f[e] - nm);
        s[g].f[e] = p;
        sum += p;
      }
      sum += __shfl_xor(sum, 1, 32);
      sum += __shfl_xor(sum, 2, 32);
      sum += __shfl_xor(sum, 4, 32);
      sum += __shfl_xor(sum, 8, 32);
      row_l[e] = row_l[e] * corr + sum;
#pragma unroll
      for (int g = 0; g < 4; ++g) acc[g].f[e] *= corr;
    }

    // bounce P through LDS to convert C-layout -> A-layout (as bf16)
    const int prow = (lane >> 4) << 3;
#pragma unroll
    for (int g = 0; g < 4; ++g)
#pragma unroll
      for (int e = 0; e < 8; ++e)
        sP[wave][prow + e][(g << 4) + (lane & 15)] = f2bf(s[g].f[e]);
    __syncthreads();

#pragma unroll
    for (int c = 0; c < 2; ++c) {
      AF pa;
      {
        const unsigned short* pb = &sP[wave][lane & 15][32 * c + ((lane >> 4) << 3)];
        pa.q[0] = *(const v4u*)pb;           // K 0..7 (+8 hi lanes)
        pa.q[1] = *(const v4u*)(pb + 16);    // K 16..23
      }
#pragma unroll
      for (int g = 0; g < 4; ++g) {
        AF bfr;
        const int nd = (g << 4) + (lane & 15);      // d column
        const unsigned short* vb = &Vt[nd][32 * c + ((lane >> 4) << 3)];
        bfr.q[0] = *(const v4u*)vb;
        bfr.q[1] = *(const v4u*)(vb + 16);
        acc[g].v = __builtin_amdgcn_wmma_f32_16x16x32_bf16(
            false, pa.v, false, bfr.v, (short)0, acc[g].v, false, false);
      }
    }

#if USE_TDM
    if (wave == 0) __builtin_amdgcn_s_wait_tensorcnt(0);  // next raw tile landed
#endif
    __syncthreads();   // publish next buffer / release sP, sKq, current tiles
  }

  const int rbase = n0 + (wave << 4) + ((lane >> 4) << 3);
#pragma unroll
  for (int e = 0; e < 8; ++e) {
    const float inv = 1.0f / row_l[e];
    const int n = rbase + e;
#pragma unroll
    for (int g = 0; g < 4; ++g) {
      const int d = (g << 4) + (lane & 15);
      Xout[base + (size_t)(d * NHEAD + h) * SEQ_N + n] = acc[g].f[e] * inv;
    }
  }
}

extern "C" void kernel_launch(void* const* d_in, const int* in_sizes, int n_in,
                              void* d_out, int out_size, void* d_ws, size_t ws_size,
                              hipStream_t stream) {
  (void)in_sizes; (void)n_in; (void)out_size; (void)ws_size;
  const float* q  = (const float*)d_in[0];
  const float* k  = (const float*)d_in[1];
  const float* v  = (const float*)d_in[2];
  const float* Wq = (const float*)d_in[3];
  const float* bq = (const float*)d_in[4];
  const float* Wk = (const float*)d_in[5];
  const float* bk = (const float*)d_in[6];
  const float* Wv = (const float*)d_in[7];
  const float* bv = (const float*)d_in[8];
  const float* Wm = (const float*)d_in[9];
  const float* bm = (const float*)d_in[10];

  const size_t per = (size_t)NBATCH * D_MODEL * SEQ_N;   // elements
  unsigned short* Qp = (unsigned short*)d_ws;            // 8 MB bf16
  unsigned short* Kp = Qp + per;                         // 8 MB
  unsigned short* Vp = Kp + per;                         // 8 MB
  float*          Xa = (float*)(Vp + per);               // 16 MB f32

  dim3 blk(128);
  dim3 gproj(SEQ_N / 64, D_MODEL / 64, NBATCH);
  proj_wmma<true><<<gproj, blk, 0, stream>>>(Wq, q, bq, (void*)Qp);
  proj_wmma<true><<<gproj, blk, 0, stream>>>(Wk, k, bk, (void*)Kp);
  proj_wmma<true><<<gproj, blk, 0, stream>>>(Wv, v, bv, (void*)Vp);

  dim3 gattn(SEQ_N / 64, NHEAD, NBATCH);
  attn_wmma<<<gattn, blk, 0, stream>>>(Qp, Kp, Vp, Xa);

  proj_wmma<false><<<gproj, blk, 0, stream>>>(Wm, Xa, bm, d_out);
}
